// MultiHeadAttention_1812476199709
// MI455X (gfx1250) — matrix operand order repaired and checked by reading, not compiled, not run on hardware
//
#include <hip/hip_runtime.h>

// MultiHeadAttention forward for MI455X (gfx1250, wave32, WMMA f16->f32).
// K0 Wo -> f16
// K1 projections (WMMA); V stored pre-transposed [n][h][d][key]
// K2 flash-style attention (WMMA, 2-pass online softmax, double-buffered
//    async global->LDS staging, coalesced b128 attention stores)
// K3 fc_out (WMMA, f16 ctx staged by async copy)

#define NBATCH 8
#define SEQL   2048
#define EMB    512
#define NH     8
#define HS     64

static constexpr float ATTN_SCALE = 0.04419417382415922f;   // 1/sqrt(512)
static constexpr float MASKED_VAL = -1.0e20f * 0.04419417382415922f;

typedef __attribute__((ext_vector_type(16))) _Float16 v16h;
typedef __attribute__((ext_vector_type(8)))  _Float16 v8h;
typedef __attribute__((ext_vector_type(4)))  _Float16 v4h;
typedef __attribute__((ext_vector_type(8)))  float    v8f;
typedef int v4i __attribute__((vector_size(16)));

union V16H { v16h v; _Float16 h[16]; };
union V8H  { v8h  v; _Float16 h[8];  };

// ---- gfx1250 async global->LDS copy (ASYNCcnt path), probed signature:
// (as(1) v4i* src, as(3) v4i* dst, imm offset, imm cpol) ----
#if defined(__has_builtin)
#if __has_builtin(__builtin_amdgcn_global_load_async_to_lds_b128) && \
    __has_builtin(__builtin_amdgcn_s_wait_asynccnt)
#define USE_ASYNC_LDS 1
#endif
#endif

#if defined(USE_ASYNC_LDS)
__device__ inline void asyncCopy16(const void* g, void* l) {
  __builtin_amdgcn_global_load_async_to_lds_b128(
      (__attribute__((address_space(1))) v4i*)g,
      (__attribute__((address_space(3))) v4i*)l, 0, 0);
}
__device__ inline void asyncWait() { __builtin_amdgcn_s_wait_asynccnt(0); }
#else
__device__ inline void asyncCopy16(const void* g, void* l) {
  *(v8h*)l = *(const v8h*)g;   // 16B register copy fallback
}
__device__ inline void asyncWait() {}
#endif

__device__ inline v8f wmma_f16(v16h a, v16h b, v8f c) {
  return __builtin_amdgcn_wmma_f32_16x16x32_f16(false, a, false, b,
                                                (short)0, c, false, false);
}

// Load 8 consecutive f32 (16B-aligned, global or LDS) convert to f16.
__device__ inline void ld8f(const float* p, _Float16* d) {
  const float4* q = (const float4*)p;
  float4 a = q[0], b = q[1];
  d[0] = (_Float16)a.x; d[1] = (_Float16)a.y; d[2] = (_Float16)a.z; d[3] = (_Float16)a.w;
  d[4] = (_Float16)b.x; d[5] = (_Float16)b.y; d[6] = (_Float16)b.z; d[7] = (_Float16)b.w;
}

// A-fragment (16x32 f16): lane row = lane%16; element e -> kc = half*16 + e.
// rowp points at this lane's row.
__device__ inline v16h aFragF32(const float* rowp, int halfid) {
  V16H u;
  ld8f(rowp + halfid * 16,     u.h);
  ld8f(rowp + halfid * 16 + 8, u.h + 8);
  return u.v;
}
__device__ inline v16h aFragH(const _Float16* rowp, int halfid) {
  V16H u;
  *(v8h*)(u.h)     = *(const v8h*)(rowp + halfid * 16);
  *(v8h*)(u.h + 8) = *(const v8h*)(rowp + halfid * 16 + 8);
  return u.v;
}

// B-fragment (32x16 f16): lane col = lane%16; element e -> kc = half*16 + e.
__device__ inline v16h bFragF32(const float* p) {
  V16H u; ld8f(p, u.h); ld8f(p + 8, u.h + 8); return u.v;
}
__device__ inline v16h bFragH(const _Float16* p) {
  V16H u;
  *(v8h*)(u.h)     = *(const v8h*)p;
  *(v8h*)(u.h + 8) = *(const v8h*)(p + 8);
  return u.v;
}

// ---------------------------------------------------------------- K0: Wo -> f16
__global__ __launch_bounds__(256) void mha_cvtw_kernel(
    const float* __restrict__ Wo, _Float16* __restrict__ woh) {
  int i = (blockIdx.x * 256 + threadIdx.x) * 4;   // 256 blocks cover 512*512
  float4 x = *(const float4*)(Wo + i);
  union { v4h v; _Float16 h[4]; } u;
  u.h[0] = (_Float16)x.x; u.h[1] = (_Float16)x.y;
  u.h[2] = (_Float16)x.z; u.h[3] = (_Float16)x.w;
  *(v4h*)(woh + i) = u.v;
}

// ---------------------------------------------------------------- K1: q/k/v projections
// One wave per (n, h, 16-row seq tile). q,k stored [n][h][s][d]; v stored
// transposed [n][h][d][s] so K2 can stage it with plain async b128 copies.
__global__ __launch_bounds__(256) void mha_proj_kernel(
    const float* __restrict__ vals, const float* __restrict__ keys,
    const float* __restrict__ qry,
    const float* __restrict__ Wv, const float* __restrict__ Wk,
    const float* __restrict__ Wq,
    _Float16* __restrict__ qh, _Float16* __restrict__ kh, _Float16* __restrict__ vhT) {
  __shared__ _Float16 st[8 * 16 * HS];              // per-wave 16x64 repack tile
  int lane = threadIdx.x & 31;
  int task = blockIdx.x * 8 + (threadIdx.x >> 5);   // 8192 wave tasks
  int s16 = task & 127;
  int h   = (task >> 7) & 7;
  int n   = task >> 10;
  int sb  = s16 * 16;
  int row = lane & 15, halfid = lane >> 4;
  _Float16* myT = &st[(threadIdx.x >> 5) * 16 * HS];

  const float* xs[3]   = {qry, keys, vals};
  const float* Ws[3]   = {Wq, Wk, Wv};
  _Float16*    outs[3] = {qh, kh, vhT};

#pragma unroll
  for (int p = 0; p < 3; ++p) {
    const float* xrow = xs[p] + ((size_t)n * SEQL + (sb + row)) * EMB + h * HS;
    v16h a0 = aFragF32(xrow, halfid);
    v16h a1 = aFragF32(xrow + 32, halfid);
#pragma unroll
    for (int dt = 0; dt < 4; ++dt) {
      int dout = dt * 16 + row;
      v16h b0 = bFragF32(Ws[p] + dout * HS + halfid * 16);
      v16h b1 = bFragF32(Ws[p] + dout * HS + 32 + halfid * 16);
      v8f c = {0.f, 0.f, 0.f, 0.f, 0.f, 0.f, 0.f, 0.f};
      c = wmma_f16(a0, b0, c);
      c = wmma_f16(a1, b1, c);
#pragma unroll
      for (int r = 0; r < 8; ++r)
        myT[(r + 8 * halfid) * HS + dt * 16 + row] = (_Float16)c[r];
    }
    if (p == 2) {
      // transposed store: vhT[((n*NH+h)*HS + d)*SEQL + sb + s]
#pragma unroll
      for (int j = 0; j < 4; ++j) {
        int u = lane + 32 * j;        // 0..127 units of (d, 8-seq half)
        int d  = u >> 1;
        int sh = (u & 1) * 8;
        V8H t;
#pragma unroll
        for (int i = 0; i < 8; ++i) t.h[i] = myT[(sh + i) * HS + d];
        *(v8h*)(vhT + (((size_t)n * NH + h) * HS + d) * SEQL + sb + sh) = t.v;
      }
    } else {
      _Float16* outp = outs[p] + (((size_t)n * NH + h) * SEQL + sb) * HS;
#pragma unroll
      for (int j = 0; j < 4; ++j)
        *(v8h*)(outp + lane * 32 + j * 8) = *(const v8h*)(myT + lane * 32 + j * 8);
    }
  }
}

// ---------------------------------------------------------------- K2: attention core
// Block = 8 waves, one (n,h), 128 q rows (16 per wave).
__global__ __launch_bounds__(256) void mha_attn_kernel(
    const _Float16* __restrict__ qh, const _Float16* __restrict__ kh,
    const _Float16* __restrict__ vT_g, const int* __restrict__ mask,
    float* __restrict__ attnOut, _Float16* __restrict__ ctxh) {
  constexpr int CH  = 32;    // keys per chunk
  constexpr int VST = 40;    // vT row stride (16B-aligned async dests)
  __shared__ _Float16 kk[2][CH * HS];     // [key][d], double buffered
  __shared__ _Float16 vT[2][HS * VST];    // [d][key], double buffered
  __shared__ float    pf[8 * 16 * CH];    // per-wave f32 P tiles (16x32)

  int tid = threadIdx.x;
  int lane = tid & 31, w = tid >> 5;
  int b = blockIdx.x;
  int q128 = b & 15, h = (b >> 4) & 7, n = b >> 7;
  int qb = q128 * 128 + w * 16;
  int row = lane & 15, halfid = lane >> 4;

  const _Float16* qbase = qh + (((size_t)n * NH + h) * SEQL) * HS;
  const _Float16* kbase = kh + (((size_t)n * NH + h) * SEQL) * HS;
  const _Float16* vbase = vT_g + (((size_t)n * NH + h) * HS) * SEQL;  // [d][key]
  const int* mrow = mask + n * SEQL;

  // per-thread staging slices
  int vd   = tid >> 2;              // 0..63
  int voff = (tid & 3) * 8;         // 0,8,16,24
  const _Float16* vsrc = vbase + (size_t)vd * SEQL + voff;
  _Float16* vdst0 = &vT[0][vd * VST + voff];
  _Float16* vdst1 = &vT[1][vd * VST + voff];

  const _Float16* qrow = qbase + (size_t)(qb + row) * HS;
  v16h qa0 = aFragH(qrow, halfid);        // d 0..31
  v16h qa1 = aFragH(qrow + 32, halfid);   // d 32..63

  float m[8], ssum[8];
#pragma unroll
  for (int r = 0; r < 8; ++r) { m[r] = -3.0e38f; ssum[r] = 0.f; }

  // ---- Pass A: online softmax statistics (K only, double buffered)
  asyncCopy16(&kbase[tid * 8], &kk[0][tid * 8]);
  asyncWait();
  __syncthreads();
  for (int kb = 0; kb < SEQL; kb += CH) {
    int ibuf = (kb >> 5) & 1;
    if (kb + CH < SEQL)
      asyncCopy16(&kbase[(size_t)(kb + CH) * HS + tid * 8], &kk[ibuf ^ 1][tid * 8]);
#pragma unroll
    for (int kt = 0; kt < 2; ++kt) {
      int msk = mrow[kb + kt * 16 + row];
      const _Float16* krowp = &kk[ibuf][(kt * 16 + row) * HS];
      v8f c = {0.f, 0.f, 0.f, 0.f, 0.f, 0.f, 0.f, 0.f};
      c = wmma_f16(qa0, bFragH(krowp + halfid * 16), c);
      c = wmma_f16(qa1, bFragH(krowp + 32 + halfid * 16), c);
#pragma unroll
      for (int r = 0; r < 8; ++r) {
        float v  = msk ? c[r] * ATTN_SCALE : MASKED_VAL;
        float nm = fmaxf(m[r], v);
        ssum[r]  = ssum[r] * __expf(m[r] - nm) + __expf(v - nm);
        m[r]     = nm;
      }
    }
    asyncWait();
    __syncthreads();
  }
  // Cross-lane combine within each 16-lane half (rows never cross halves).
#pragma unroll
  for (int r = 0; r < 8; ++r) {
    float mm = m[r], s = ssum[r];
#pragma unroll
    for (int o = 1; o < 16; o <<= 1) {
      float mo = __shfl_xor(mm, o, 16);
      float so = __shfl_xor(s, o, 16);
      float M  = fmaxf(mm, mo);
      s  = s * __expf(mm - M) + so * __expf(mo - M);
      mm = M;
    }
    m[r] = mm;
    ssum[r] = 1.0f / s;   // inverse denominator
  }

  // ---- Pass B: normalized attention + PV (K and V, double buffered)
  v8f cacc[4];
#pragma unroll
  for (int dt = 0; dt < 4; ++dt) cacc[dt] = (v8f){0.f,0.f,0.f,0.f,0.f,0.f,0.f,0.f};
  float* arow = attnOut + (((size_t)n * NH + h) * SEQL + qb) * SEQL;
  float* myPF = &pf[w * 16 * CH];
  int prRow = lane >> 1;            // repack: row 0..15
  int prCol = (lane & 1) * 16;      // repack: col 0 or 16

  asyncCopy16(&kbase[tid * 8], &kk[0][tid * 8]);
  asyncCopy16(vsrc, vdst0);
  asyncWait();
  __syncthreads();
  for (int kb = 0; kb < SEQL; kb += CH) {
    int ibuf = (kb >> 5) & 1;
    if (kb + CH < SEQL) {
      asyncCopy16(&kbase[(size_t)(kb + CH) * HS + tid * 8], &kk[ibuf ^ 1][tid * 8]);
      asyncCopy16(vsrc + kb + CH, ibuf ? vdst0 : vdst1);
    }
#pragma unroll
    for (int kt = 0; kt < 2; ++kt) {
      int msk = mrow[kb + kt * 16 + row];
      const _Float16* krowp = &kk[ibuf][(kt * 16 + row) * HS];
      v8f c = {0.f, 0.f, 0.f, 0.f, 0.f, 0.f, 0.f, 0.f};
      c = wmma_f16(qa0, bFragH(krowp + halfid * 16), c);
      c = wmma_f16(qa1, bFragH(krowp + 32 + halfid * 16), c);
#pragma unroll
      for (int r = 0; r < 8; ++r) {
        float v = msk ? c[r] * ATTN_SCALE : MASKED_VAL;
        float p = __expf(v - m[r]) * ssum[r];
        myPF[(r + 8 * halfid) * CH + kt * 16 + row] = p;   // wave-private LDS
      }
    }
    // Coalesced attention output: each lane stores 16 f32 (4 x b128).
    {
      const float4* ps = (const float4*)&myPF[prRow * CH + prCol];
      float4* gd = (float4*)(arow + (size_t)prRow * SEQL + kb + prCol);
      float4 x0 = ps[0], x1 = ps[1], x2 = ps[2], x3 = ps[3];
      gd[0] = x0; gd[1] = x1; gd[2] = x2; gd[3] = x3;
    }
    // PV: A = P tile (16x32 keys, f32->f16 at frag build), B = V tiles
    v16h pa = aFragF32(&myPF[row * CH], halfid);
#pragma unroll
    for (int dt = 0; dt < 4; ++dt) {
      v16h vb = bFragH(&vT[ibuf][(dt * 16 + row) * VST + halfid * 16]);
      cacc[dt] = wmma_f16(pa, vb, cacc[dt]);
    }
    asyncWait();
    __syncthreads();
  }

  // ctx (f16) -> ws, layout [n][q][h*64+d]; repack via wave-private LDS.
  _Float16* cst = (_Float16*)myPF;                  // 16x64 f16 (2 KB)
#pragma unroll
  for (int dt = 0; dt < 4; ++dt)
#pragma unroll
    for (int r = 0; r < 8; ++r)
      cst[(r + 8 * halfid) * HS + dt * 16 + row] = (_Float16)cacc[dt][r];
  {
    int cr = lane >> 1, coff = (lane & 1) * 32;
    _Float16* gdst = ctxh + ((size_t)n * SEQL + qb + cr) * EMB + h * HS + coff;
#pragma unroll
    for (int j = 0; j < 4; ++j)
      *(v8h*)(gdst + j * 8) = *(const v8h*)(cst + cr * HS + coff + j * 8);
  }
}

// ---------------------------------------------------------------- K3: fc_out
__global__ __launch_bounds__(256) void mha_fcout_kernel(
    const _Float16* __restrict__ woh, const float* __restrict__ bo,
    const _Float16* __restrict__ ctxh, float* __restrict__ out) {
  __shared__ _Float16 cth[16 * EMB];          // 16 KB f16 ctx tile
  int tid = threadIdx.x, lane = tid & 31, w = tid >> 5;
  int b = blockIdx.x;
  int s16 = b & 127, n = b >> 7;
  int sb = s16 * 16;
  const _Float16* cbase = ctxh + ((size_t)n * SEQL + sb) * EMB;

#pragma unroll
  for (int j = 0; j < 4; ++j)                 // 4 x 16B per thread, coalesced
    asyncCopy16(cbase + (j * 256 + tid) * 8, cth + (j * 256 + tid) * 8);
  asyncWait();
  __syncthreads();

  int row = lane & 15, halfid = lane >> 4;
  v8f acc[4];
#pragma unroll
  for (int dt = 0; dt < 4; ++dt) acc[dt] = (v8f){0.f,0.f,0.f,0.f,0.f,0.f,0.f,0.f};

  for (int ks = 0; ks < 16; ++ks) {           // contraction chunks of 32 over E
    v16h a = aFragH(&cth[row * EMB + ks * 32], halfid);
#pragma unroll
    for (int dt = 0; dt < 4; ++dt) {
      int dout = (w * 4 + dt) * 16 + row;
      v16h bf = bFragH(woh + (size_t)dout * EMB + ks * 32 + halfid * 16);
      acc[dt] = wmma_f16(a, bf, acc[dt]);
    }
  }
  float* base = out + ((size_t)n * SEQL + sb) * EMB;
#pragma unroll
  for (int dt = 0; dt < 4; ++dt) {
    int dout = (w * 4 + dt) * 16 + row;
    float bias = bo[dout];
#pragma unroll
    for (int r = 0; r < 8; ++r)
      base[(size_t)(r + 8 * halfid) * EMB + dout] = acc[dt][r] + bias;
  }
}

// ----------------------------------------------------------------
extern "C" void kernel_launch(void* const* d_in, const int* in_sizes, int n_in,
                              void* d_out, int out_size, void* d_ws, size_t ws_size,
                              hipStream_t stream) {
  const float* vals = (const float*)d_in[0];
  const float* keys = (const float*)d_in[1];
  const float* qry  = (const float*)d_in[2];
  const int*   mask = (const int*)d_in[3];
  const float* Wv   = (const float*)d_in[4];
  const float* Wk   = (const float*)d_in[5];
  const float* Wq   = (const float*)d_in[6];
  const float* Wo   = (const float*)d_in[7];
  const float* bo   = (const float*)d_in[8];

  float* out  = (float*)d_out;                               // [8,2048,512]
  float* attn = out + (size_t)NBATCH * SEQL * EMB;           // [8,8,2048,2048]

  // Workspace (f16): q, k, vT, ctx (16 MiB each) + Wo (512 KiB) = ~64.5 MiB.
  size_t per = (size_t)NBATCH * NH * SEQL * HS;
  _Float16* qh   = (_Float16*)d_ws;
  _Float16* kh   = qh + per;
  _Float16* vhT  = kh + per;
  _Float16* ctxh = vhT + per;
  _Float16* woh  = ctxh + per;

  mha_cvtw_kernel <<<256, 256, 0, stream>>>(Wo, woh);
  mha_proj_kernel <<<1024, 256, 0, stream>>>(vals, keys, qry, Wv, Wk, Wq, qh, kh, vhT);
  mha_attn_kernel <<<1024, 256, 0, stream>>>(qh, kh, vhT, mask, attn, ctxh);
  mha_fcout_kernel<<<1024, 256, 0, stream>>>(woh, bo, ctxh, out);
}
